// Decode_85375359910656
// MI455X (gfx1250) — matrix operand order repaired and checked by reading, not compiled, not run on hardware
//
#include <hip/hip_runtime.h>
#include <hip/hip_bf16.h>

typedef _Float16 v16h __attribute__((ext_vector_type(16)));
typedef _Float16 v8h  __attribute__((ext_vector_type(8)));
typedef _Float16 v2h  __attribute__((ext_vector_type(2)));
typedef float    v8f  __attribute__((ext_vector_type(8)));

static __device__ __forceinline__ v8f wmma16(v16h a, v16h b, v8f c) {
  // D = A(16x32 f16) * B(32x16 f16) + C(16x16 f32)
  return __builtin_amdgcn_wmma_f32_16x16x32_f16(false, a, false, b, (short)0, c, false, false);
}
static __device__ __forceinline__ v16h cat16(v8h lo, v8h hi) {
  return __builtin_shufflevector(lo, hi, 0,1,2,3,4,5,6,7,8,9,10,11,12,13,14,15);
}
static __device__ __forceinline__ int imin(int a, int b){ return a < b ? a : b; }
static __device__ __forceinline__ int imax(int a, int b){ return a > b ? a : b; }

#define HH 256
#define WW 256
#define CIN 64
#define CMID 256
#define COUT 64
#define NPTS 129
#define KPOLY 8256   /* 129*64 = 258 k-steps of 32 */
#define NPOLY 512
#define KFUSE 512
#define NFUSE 256

// ---------------- NCHW fp32 -> NHWC f16 (c-minor, 64ch/pixel) ----------------
__global__ __launch_bounds__(256) void k_convert_in(const float* __restrict__ src,
                                                    _Float16* __restrict__ dst) {
  int bid = blockIdx.x;                 // 4 imgs * 256 rows * 4 xtiles
  int xt = bid & 3, y = (bid >> 2) & 255, img = bid >> 10;
  int x0 = xt * 64;
  __shared__ float tile[64][65];
  int t = threadIdx.x;
  for (int i = 0; i < 16; i++) {
    int e = t + i * 256; int c = e >> 6, x = e & 63;
    tile[c][x] = src[(((size_t)(img * 64 + c) * HH + y) * WW) + x0 + x];
  }
  __syncthreads();
  int x = t >> 2, cb = (t & 3) * 16;
  _Float16* o = dst + ((size_t)((img * HH + y) * WW) + x0 + x) * 64 + cb;
  v8h a, b;
  for (int i = 0; i < 8; i++) a[i] = (_Float16)tile[cb + i][x];
  for (int i = 0; i < 8; i++) b[i] = (_Float16)tile[cb + 8 + i][x];
  *(v8h*)o = a; *((v8h*)o + 1) = b;
}

// ---------------- weight fragment packers (ISA B layout) ----------------
// frag element index = ((kk*NT + nt)*32 + lane)*16 + e ;  n = nt*16 + (lane&15)
// k = kk*32 + (lane>>4)*8 + (e&7) + (e>>3)*16   (ISA table order)
__global__ __launch_bounds__(256) void k_pack_w1(const float* __restrict__ w, _Float16* __restrict__ o) {
  int idx = blockIdx.x * 256 + threadIdx.x;
  if (idx >= 18 * 16 * 32 * 16) return;
  int e = idx & 15, lane = (idx >> 4) & 31, nt = (idx >> 9) & 15, kk = idx >> 13;
  int n = nt * 16 + (lane & 15);
  int k = kk * 32 + ((lane >> 4) * 8) + (e & 7) + ((e >> 3) * 16);      // K = 576, tap-major
  int ic = k & 63, tap = k >> 6;
  o[idx] = (_Float16)w[((n * 64 + ic) * 3 + tap / 3) * 3 + (tap % 3)];
}
__global__ __launch_bounds__(256) void k_pack_w2(const float* __restrict__ w, _Float16* __restrict__ o) {
  int idx = blockIdx.x * 256 + threadIdx.x;
  if (idx >= 8 * 4 * 32 * 16) return;
  int e = idx & 15, lane = (idx >> 4) & 31, nt = (idx >> 9) & 3, kk = idx >> 11;
  int n = nt * 16 + (lane & 15);
  int k = kk * 32 + ((lane >> 4) * 8) + (e & 7) + ((e >> 3) * 16);      // K = 256
  o[idx] = (_Float16)w[n * 256 + k];
}
__global__ __launch_bounds__(256) void k_pack_wpoly(const float* __restrict__ w, _Float16* __restrict__ o) {
  int idx = blockIdx.x * 256 + threadIdx.x;
  if (idx >= 258 * 32 * 32 * 16) return;
  int e = idx & 15, lane = (idx >> 4) & 31, nt = (idx >> 9) & 31, kk = idx >> 14;
  int n = nt * 16 + (lane & 15);
  int k = kk * 32 + ((lane >> 4) * 8) + (e & 7) + ((e >> 3) * 16);      // k' = p*64 + c (sampler order)
  int p = k >> 6, c = k & 63;
  o[idx] = (_Float16)w[(size_t)n * KPOLY + c * NPTS + p];   // orig col = c*129 + p
}
__global__ __launch_bounds__(256) void k_pack_wfuse(const float* __restrict__ w, _Float16* __restrict__ o) {
  int idx = blockIdx.x * 256 + threadIdx.x;
  if (idx >= 16 * 16 * 32 * 16) return;
  int e = idx & 15, lane = (idx >> 4) & 31, nt = (idx >> 9) & 15, kk = idx >> 13;
  int n = nt * 16 + (lane & 15);
  int k = kk * 32 + ((lane >> 4) * 8) + (e & 7) + ((e >> 3) * 16);      // K = 512
  o[idx] = (_Float16)w[n * 512 + k];
}

// ---------------- fused conv3x3(64->256)+relu+conv1x1(256->64) ----------------
__global__ __launch_bounds__(256) void k_conv_fused(
    const _Float16* __restrict__ in16, const _Float16* __restrict__ w1f,
    const _Float16* __restrict__ w2f, const float* __restrict__ b1,
    const float* __restrict__ b2, _Float16* __restrict__ fout) {
  int bid = blockIdx.x;
  int xt = bid & 3, y = (bid >> 2) & 255, img = bid >> 10;
  int x0 = xt * 64;
  int t = threadIdx.x, w = t >> 5, lane = t & 31;
  int l15 = lane & 15, lhi = lane >> 4;

  __shared__ _Float16 mid[64][CMID];      // 32 KB: relu(conv1) tile, f16
  __shared__ _Float16 obuf[64 * COUT];    //  8 KB: conv2 output staging

  // ---- stage 1: M=64 pixels, N=256 mid-ch, K=576 (9 taps x 64 in-ch) ----
  int mt = w & 3, nh = w >> 2;            // wave -> Mtile(0..3), N-half(0..1)
  v8f acc[8] = {};
  int m = mt * 16 + l15;
  for (int kk = 0; kk < 18; kk++) {
    int tap = kk >> 1;
    int cb = (kk & 1) * 32 + lhi * 8;     // in-channel base for this lane group
    int yy = y + tap / 3 - 1, xx = x0 + m + (tap % 3) - 1;
    v8h alo = {}, ahi = {};
    if (yy >= 0 && yy < HH && xx >= 0 && xx < WW) {
      const _Float16* p = in16 + ((size_t)(img * HH + yy) * WW + xx) * 64 + cb;
      alo = *(const v8h*)p;               // K [base, base+8)
      ahi = *(const v8h*)(p + 16);        // K [base+16, base+24)
    }
    v16h a = cat16(alo, ahi);
    const v16h* bp = (const v16h*)w1f + ((size_t)kk * 16 + nh * 8) * 32 + lane;
#pragma unroll
    for (int j = 0; j < 8; j++) acc[j] = wmma16(a, bp[j * 32], acc[j]);
  }
#pragma unroll
  for (int j = 0; j < 8; j++) {
    int n = (nh * 8 + j) * 16 + l15;
    float bias = b1[n];
#pragma unroll
    for (int r = 0; r < 8; r++) {
      float v = acc[j][r] + bias;
      mid[mt * 16 + lhi * 8 + r][n] = (_Float16)(v > 0.f ? v : 0.f);
    }
  }
  __syncthreads();

  // ---- stage 2: M=64 pixels, N=64 out-ch, K=256 (from LDS) ----
  int mt2 = w & 3, nb2 = (w >> 2) * 2;    // wave -> Mtile, 2 Ntiles
  v8f acc2[2] = {};
  int m2 = mt2 * 16 + l15;
  for (int kk = 0; kk < 8; kk++) {
    int kb = kk * 32 + lhi * 8;
    v8h alo = *(const v8h*)&mid[m2][kb];
    v8h ahi = *(const v8h*)&mid[m2][kb + 16];
    v16h a = cat16(alo, ahi);
#pragma unroll
    for (int j = 0; j < 2; j++) {
      v16h bf = *((const v16h*)w2f + ((size_t)kk * 4 + nb2 + j) * 32 + lane);
      acc2[j] = wmma16(a, bf, acc2[j]);
    }
  }
#pragma unroll
  for (int j = 0; j < 2; j++) {
    int n = (nb2 + j) * 16 + l15;
    float bias = b2[n];
#pragma unroll
    for (int r = 0; r < 8; r++)
      obuf[(mt2 * 16 + lhi * 8 + r) * 64 + n] = (_Float16)(acc2[j][r] + bias);
  }
  __syncthreads();
  // coalesced 32B/thread store of the 64x64 NHWC tile
  _Float16* dst = fout + ((size_t)(img * HH + y) * WW + x0) * 64;
  v8h* s = (v8h*)&obuf[t * 16];
  *((v8h*)(dst + t * 16)) = s[0];
  *((v8h*)(dst + t * 16) + 1) = s[1];
}

// ---------------- gather + init_polys + bilinear sampling ----------------
__global__ __launch_bounds__(256) void k_sample(
    const float* __restrict__ whp, const int* __restrict__ ct_ind,
    const int* __restrict__ ct_img, const _Float16* __restrict__ fbuf,
    _Float16* __restrict__ FP, float* __restrict__ initb,
    float* __restrict__ out0, int n_inst) {
  int mI = blockIdx.x, t = threadIdx.x;
  _Float16* fprow = FP + (size_t)mI * KPOLY;
  if (mI >= n_inst) {                     // zero-pad rows for the GEMM
    v8h z = {};
    for (int i = t; i < KPOLY / 8; i += 256) *((v8h*)fprow + i) = z;
    return;
  }
  __shared__ float ptsx[NPTS], ptsy[NPTS];
  int ci = ct_ind[mI];
  int bi = ct_img[mI];
  int ctx = imin(imax(ci % WW, 0), WW - 1);
  int cty = imin(imax(ci / WW, 0), HH - 1);
  {
    float ov = whp[(((size_t)bi * 256 + t) * HH + cty) * WW + ctx];
    float ctv = (t & 1) ? (float)cty : (float)ctx;
    float init = ov * 10.0f + ctv;                 // INIT_STRIDE
    out0[(size_t)mI * 256 + t] = init * 4.0f;      // DOWN_SAMPLE
    initb[(size_t)mI * 256 + t] = init;
    if ((t & 1) == 0) ptsx[1 + (t >> 1)] = init; else ptsy[1 + (t >> 1)] = init;
  }
  if (t == 0) { ptsx[0] = (float)ctx; ptsy[0] = (float)cty; }
  __syncthreads();

  int w = t >> 5, lane = t & 31;
  int c0 = lane * 2;                      // 2 channels per lane
  for (int p = w; p < NPTS; p += 8) {
    float sx = ptsx[p] - 0.5f, sy = ptsy[p] - 0.5f;
    float fx = floorf(sx), fy = floorf(sy);
    int x0i = (int)fx, y0i = (int)fy;
    float wx1 = sx - fx, wy1 = sy - fy;
    float wx0 = 1.f - wx1, wy0 = 1.f - wy1;
    float a0 = 0.f, a1 = 0.f;
#pragma unroll
    for (int cor = 0; cor < 4; cor++) {
      int xc = x0i + (cor & 1), yc = y0i + (cor >> 1);
      float wgt = ((cor & 1) ? wx1 : wx0) * ((cor >> 1) ? wy1 : wy0);
      bool valid = (xc >= 0) && (xc < WW) && (yc >= 0) && (yc < HH);
      wgt = valid ? wgt : 0.f;
      int xcc = imin(imax(xc, 0), WW - 1), ycc = imin(imax(yc, 0), HH - 1);
      v2h v = *(const v2h*)(fbuf + ((size_t)(bi * HH + ycc) * WW + xcc) * 64 + c0);
      a0 += wgt * (float)v[0];
      a1 += wgt * (float)v[1];
    }
    v2h r; r[0] = (_Float16)a0; r[1] = (_Float16)a1;
    *(v2h*)(fprow + p * 64 + c0) = r;     // K order = p*64 + c (matches wpoly pack)
  }
}

// ---------------- GEMM1: (Mpad x 8256) x (8256 x 512) -> f16 ----------------
__global__ __launch_bounds__(256) void k_gemm1(const _Float16* __restrict__ A,
                                               const _Float16* __restrict__ Bf,
                                               _Float16* __restrict__ C) {
  int t = threadIdx.x, w = t >> 5, lane = t & 31, l15 = lane & 15, lhi = lane >> 4;
  int mbase = blockIdx.x * 64, nb = blockIdx.y;          // nb 0..3 (128 cols each)
  int mt = w & 3, nh = w >> 2;
  const _Float16* arow = A + (size_t)(mbase + mt * 16 + l15) * KPOLY;
  int ntb = nb * 8 + nh * 4;
  v8f acc[4] = {};
  for (int kk = 0; kk < 258; kk++) {
    int kb = kk * 32 + lhi * 8;
    v16h a = cat16(*(const v8h*)(arow + kb), *(const v8h*)(arow + kb + 16));
    const v16h* bp = (const v16h*)Bf + ((size_t)kk * 32 + ntb) * 32 + lane;
#pragma unroll
    for (int j = 0; j < 4; j++) acc[j] = wmma16(a, bp[j * 32], acc[j]);
  }
#pragma unroll
  for (int j = 0; j < 4; j++) {
    int n = (ntb + j) * 16 + l15;
#pragma unroll
    for (int r = 0; r < 8; r++)
      C[(size_t)(mbase + mt * 16 + lhi * 8 + r) * NPOLY + n] = (_Float16)acc[j][r];
  }
}

// ---------------- GEMM2 + epilogue: out = off*16 + init*4 ----------------
__global__ __launch_bounds__(256) void k_gemm2(const _Float16* __restrict__ A,
                                               const _Float16* __restrict__ Bf,
                                               const float* __restrict__ bfuse,
                                               const float* __restrict__ initb,
                                               float* __restrict__ out1, int n_inst) {
  int t = threadIdx.x, w = t >> 5, lane = t & 31, l15 = lane & 15, lhi = lane >> 4;
  int mbase = blockIdx.x * 64, nb = blockIdx.y;          // nb 0..1
  int mt = w & 3, nh = w >> 2;
  const _Float16* arow = A + (size_t)(mbase + mt * 16 + l15) * KFUSE;
  int ntb = nb * 8 + nh * 4;
  v8f acc[4] = {};
  for (int kk = 0; kk < 16; kk++) {
    int kb = kk * 32 + lhi * 8;
    v16h a = cat16(*(const v8h*)(arow + kb), *(const v8h*)(arow + kb + 16));
    const v16h* bp = (const v16h*)Bf + ((size_t)kk * 16 + ntb) * 32 + lane;
#pragma unroll
    for (int j = 0; j < 4; j++) acc[j] = wmma16(a, bp[j * 32], acc[j]);
  }
#pragma unroll
  for (int j = 0; j < 4; j++) {
    int n = (ntb + j) * 16 + l15;
    float bias = bfuse[n];
#pragma unroll
    for (int r = 0; r < 8; r++) {
      int mm = mbase + mt * 16 + lhi * 8 + r;
      if (mm < n_inst) {
        float off = acc[j][r] + bias;
        // coarse*4 = (off*COARSE_STRIDE + init)*DOWN_SAMPLE = off*16 + init*4
        out1[(size_t)mm * 256 + n] = off * 16.0f + initb[(size_t)mm * 256 + n] * 4.0f;
      }
    }
  }
}

extern "C" void kernel_launch(void* const* d_in, const int* in_sizes, int n_in,
                              void* d_out, int out_size, void* d_ws, size_t ws_size,
                              hipStream_t stream) {
  (void)n_in; (void)out_size; (void)ws_size;
  const float* cnn   = (const float*)d_in[0];
  const float* whp   = (const float*)d_in[1];
  const int*   ctind = (const int*)d_in[2];
  const int*   ctimg = (const int*)d_in[3];
  const float* w1    = (const float*)d_in[4];
  const float* b1    = (const float*)d_in[5];
  const float* w2    = (const float*)d_in[6];
  const float* b2    = (const float*)d_in[7];
  const float* wpoly = (const float*)d_in[8];
  const float* wfuse = (const float*)d_in[9];
  const float* bfuse = (const float*)d_in[10];
  int n_inst = in_sizes[2];
  int Mpad = ((n_inst + 63) / 64) * 64;

  char* ws = (char*)d_ws;
  size_t off = 0;
  auto alloc = [&](size_t bytes) -> char* {
    char* p = ws + off; off += (bytes + 255) & ~(size_t)255; return p;
  };
  _Float16* in16 = (_Float16*)alloc((size_t)4 * 256 * 256 * 64 * 2);
  _Float16* fbuf = (_Float16*)alloc((size_t)4 * 256 * 256 * 64 * 2);
  _Float16* FP   = (_Float16*)alloc((size_t)Mpad * KPOLY * 2);
  _Float16* OFF1 = (_Float16*)alloc((size_t)Mpad * NPOLY * 2);
  float*    inib = (float*)alloc((size_t)Mpad * 256 * 4);
  _Float16* w1f  = (_Float16*)alloc((size_t)18 * 16 * 32 * 16 * 2);
  _Float16* w2f  = (_Float16*)alloc((size_t)8 * 4 * 32 * 16 * 2);
  _Float16* wpf  = (_Float16*)alloc((size_t)258 * 32 * 32 * 16 * 2);
  _Float16* wff  = (_Float16*)alloc((size_t)16 * 16 * 32 * 16 * 2);

  float* out0 = (float*)d_out;
  float* out1 = out0 + (size_t)n_inst * 256;

  k_pack_w1   <<<(18 * 16 * 32 * 16) / 256, 256, 0, stream>>>(w1, w1f);
  k_pack_w2   <<<(8 * 4 * 32 * 16) / 256, 256, 0, stream>>>(w2, w2f);
  k_pack_wpoly<<<(258 * 32 * 32 * 16) / 256, 256, 0, stream>>>(wpoly, wpf);
  k_pack_wfuse<<<(16 * 16 * 32 * 16) / 256, 256, 0, stream>>>(wfuse, wff);
  k_convert_in<<<4096, 256, 0, stream>>>(cnn, in16);
  k_conv_fused<<<4096, 256, 0, stream>>>(in16, w1f, w2f, b1, b2, fbuf);
  k_sample    <<<Mpad, 256, 0, stream>>>(whp, ctind, ctimg, fbuf, FP, inib, out0, n_inst);
  k_gemm1     <<<dim3(Mpad / 64, 4), 256, 0, stream>>>(FP, wpf, OFF1);
  k_gemm2     <<<dim3(Mpad / 64, 2), 256, 0, stream>>>(OFF1, wff, bfuse, inib, out1, n_inst);
}
